// Activation_83245056131825
// MI455X (gfx1250) — compile-verified, matrix operand order checked by reading, not run on hardware
//
#include <hip/hip_runtime.h>

// ---- problem dims ----
#define B_   1024
#define S_   200
#define D_   512
#define G_   1536   // 3*D
#define F_   256    // D_FF

typedef __attribute__((ext_vector_type(16))) __bf16 v16bf;
typedef __attribute__((ext_vector_type(8)))  float  v8f;

using u16 = unsigned short;
using u32 = unsigned int;

// ---- bf16 helpers (RNE-ish round) ----
__device__ __forceinline__ u16 f2bf(float f) {
  u32 u = __builtin_bit_cast(u32, f);
  u = u + 0x7FFFu + ((u >> 16) & 1u);
  return (u16)(u >> 16);
}
__device__ __forceinline__ float bf2f(u16 h) {
  u32 u = ((u32)h) << 16;
  return __builtin_bit_cast(float, u);
}

// ---- CDNA5 async global->LDS copy (ASYNCcnt path, no VGPR round-trip) ----
// Per-lane: copies 16 bytes from global vaddr to LDS byte offset in vdst VGPR.
__device__ __forceinline__ void async_copy_b128(u32 lds_byte_off, const void* gsrc) {
  asm volatile("global_load_async_to_lds_b128 %0, %1, off"
               :: "v"(lds_byte_off), "v"(gsrc)
               : "memory");
}
__device__ __forceinline__ void wait_async_all() {
  asm volatile("s_wait_asynccnt 0x0" ::: "memory");
}

// ---- WMMA bf16 fragment load (A or B operand) ----
// 16-bit operand layout (CDNA5 ISA 7.12.2): lane&15 = row (A:M / B:N),
// half = lane>>4 selects K octet. VGPR v (0..7) holds K pair
// (v>>2)*16 + half*8 + (v&3)*2.  So elements 0..7  = K0+half*8 .. +7,
// elements 8..15 = K0+16+half*8 .. +7 -> two contiguous 16B loads.
__device__ __forceinline__ v16bf load_frag(const u16* base, int ld, int row0, int k0) {
  const int lane = threadIdx.x & 31;
  const int idx  = lane & 15;
  const int half = lane >> 4;
  const u16* p = base + (size_t)(row0 + idx) * ld + k0 + half * 8;
  uint4 lo = *reinterpret_cast<const uint4*>(p);
  uint4 hi = *reinterpret_cast<const uint4*>(p + 16);
  v16bf f;
  reinterpret_cast<uint4*>(&f)[0] = lo;
  reinterpret_cast<uint4*>(&f)[1] = hi;
  return f;
}

__device__ __forceinline__ v8f wmma_bf16(v16bf a, v16bf b, v8f c) {
  // (neg_a, A, neg_b, B, c_mod, C, reuse_a, reuse_b)
  return __builtin_amdgcn_wmma_f32_16x16x32_bf16(false, a, false, b, (short)0, c, false, false);
}

// =====================================================================
// Kernel: f32 -> bf16 conversion (strided source, e.g. W_fc column split)
// =====================================================================
__global__ void conv_bf16_kernel(const float* __restrict__ src, u16* __restrict__ dst,
                                 int rows, int cols, int src_ld, int src_off) {
  int n = rows * cols;
  for (int i = blockIdx.x * blockDim.x + threadIdx.x; i < n; i += gridDim.x * blockDim.x) {
    int r = i / cols, c = i - r * cols;
    dst[i] = f2bf(src[(size_t)r * src_ld + src_off + c]);
  }
}

__global__ void zero_h_kernel(float* __restrict__ hf, u16* __restrict__ hb, int n) {
  int i = blockIdx.x * blockDim.x + threadIdx.x;
  if (i < n) { hf[i] = 0.f; hb[i] = 0; }
}

// =====================================================================
// Kernel 1: gx = history @ W_ih^T + b_ih   (M=B*S, N=G, K=D), bf16 out
// One block per 16-row M tile; 8 waves each own 12 N-tiles (96 total).
// A panel needs f32->bf16 conversion in flight -> VGPR staging path.
// =====================================================================
__global__ void __launch_bounds__(256)
gemm_gx_kernel(const float* __restrict__ history, const u16* __restrict__ Wih,
               const float* __restrict__ b_ih, u16* __restrict__ gx) {
  __shared__ __align__(16) u16 At[16 * D_];   // 16 KB A panel (converted to bf16)
  const int m0  = blockIdx.x * 16;
  const int tid = threadIdx.x;
  __builtin_prefetch(history + (size_t)m0 * D_, 0, 1);   // global_prefetch_b8
  for (int i = tid; i < 16 * D_; i += 256)
    At[i] = f2bf(history[(size_t)m0 * D_ + i]);          // rows are contiguous
  __syncthreads();

  const int wave = tid >> 5, lane = tid & 31;
  const int idx = lane & 15, half = lane >> 4;
  v8f acc[12] = {};
  for (int k = 0; k < 16; ++k) {
    v16bf a = load_frag(At, D_, 0, k * 32);
#pragma unroll
    for (int t = 0; t < 12; ++t) {
      int g0 = (wave * 12 + t) * 16;
      v16bf b = load_frag(Wih, D_, g0, k * 32);
      acc[t] = wmma_bf16(a, b, acc[t]);
    }
  }
  // D layout: VGPR r -> M = r + 8*half, N = lane&15
#pragma unroll
  for (int t = 0; t < 12; ++t) {
    int g = (wave * 12 + t) * 16 + idx;
    float bias = b_ih[g];
#pragma unroll
    for (int r = 0; r < 8; ++r) {
      int m = m0 + r + half * 8;
      gx[(size_t)m * G_ + g] = f2bf(acc[t][r] + bias);
    }
  }
}

// =====================================================================
// Kernel 2: one GRU time step.  Each block = 16 batch rows.
// h panel staged via GLOBAL_LOAD_ASYNC_TO_LDS_B128 (ASYNCcnt).
// Wave w owns the gate triplet (r,z,n) for d-columns [w*64, w*64+64):
// tiles d0, d0+512, d0+1024 -> gating entirely in registers.
// =====================================================================
__global__ void __launch_bounds__(256)
gru_step_kernel(const u16* __restrict__ gx, const u16* __restrict__ Whh,
                const float* __restrict__ b_hh, float* __restrict__ h_f32,
                u16* __restrict__ h_bf16, u16* __restrict__ gru_out, int t) {
  __shared__ __align__(16) u16 Ht[16 * D_];   // 16 KB h panel
  const int b0  = blockIdx.x * 16;
  const int tid = threadIdx.x;
  {
    const u32 lds_base = (u32)(uintptr_t)Ht;        // low 32b of generic = LDS addr
    const u16* gsrc = h_bf16 + (size_t)b0 * D_;     // 16 rows, contiguous
#pragma unroll
    for (int i = 0; i < 4; ++i) {                   // 1024 x b128 over 256 threads
      int j = tid + i * 256;
      async_copy_b128(lds_base + j * 16, gsrc + j * 8);
    }
    wait_async_all();
  }
  __syncthreads();

  const int wave = tid >> 5, lane = tid & 31;
  const int idx = lane & 15, half = lane >> 4;
  v8f aR[4] = {}, aZ[4] = {}, aN[4] = {};
  for (int k = 0; k < 16; ++k) {
    v16bf a = load_frag(Ht, D_, 0, k * 32);
#pragma unroll
    for (int q = 0; q < 4; ++q) {
      int d0 = (wave * 4 + q) * 16;
      aR[q] = wmma_bf16(a, load_frag(Whh, D_, d0,          k * 32), aR[q]);
      aZ[q] = wmma_bf16(a, load_frag(Whh, D_, D_ + d0,     k * 32), aZ[q]);
      aN[q] = wmma_bf16(a, load_frag(Whh, D_, 2 * D_ + d0, k * 32), aN[q]);
    }
  }
#pragma unroll
  for (int q = 0; q < 4; ++q) {
    int d = (wave * 4 + q) * 16 + idx;
    float bR = b_hh[d], bZ = b_hh[D_ + d], bN = b_hh[2 * D_ + d];
#pragma unroll
    for (int r = 0; r < 8; ++r) {
      int b = b0 + r + half * 8;
      size_t row = (size_t)b * S_ + t;
      float xr = bf2f(gx[row * G_ + d]);
      float xz = bf2f(gx[row * G_ + D_ + d]);
      float xn = bf2f(gx[row * G_ + 2 * D_ + d]);
      float rg = 1.f / (1.f + __expf(-(xr + aR[q][r] + bR)));
      float zg = 1.f / (1.f + __expf(-(xz + aZ[q][r] + bZ)));
      float ng = tanhf(xn + rg * (aN[q][r] + bN));
      float hp = h_f32[(size_t)b * D_ + d];
      float hn = (1.f - zg) * ng + zg * hp;
      h_f32[(size_t)b * D_ + d] = hn;
      u16 hb = f2bf(hn);
      h_bf16[(size_t)b * D_ + d] = hb;
      gru_out[row * D_ + d] = hb;
    }
  }
}

// =====================================================================
// Kernel 3: tvec[b,f] = target @ Wt^T + b_fc   (M=B, N=F, K=D), f32 out
// =====================================================================
__global__ void __launch_bounds__(256)
tvec_kernel(const u16* __restrict__ tgt_bf, const u16* __restrict__ Wt,
            const float* __restrict__ b_fc, float* __restrict__ tvec) {
  const int m0  = blockIdx.x * 16;
  const int tid = threadIdx.x;
  const int wave = tid >> 5, lane = tid & 31;
  const int idx = lane & 15, half = lane >> 4;
  v8f acc[2] = {};
  for (int k = 0; k < 16; ++k) {
    v16bf a = load_frag(tgt_bf, D_, m0, k * 32);
#pragma unroll
    for (int q = 0; q < 2; ++q) {
      int f0 = (wave * 2 + q) * 16;
      acc[q] = wmma_bf16(a, load_frag(Wt, D_, f0, k * 32), acc[q]);
    }
  }
#pragma unroll
  for (int q = 0; q < 2; ++q) {
    int f = (wave * 2 + q) * 16 + idx;
    float bias = b_fc[f];
#pragma unroll
    for (int r = 0; r < 8; ++r) {
      int m = m0 + r + half * 8;
      tvec[(size_t)m * F_ + f] = acc[q][r] + bias;
    }
  }
}

// =====================================================================
// Kernel 4: fused  pre = gru@Wg^T + tvec[b] ; hidden = gelu(pre) ;
//           scores = hidden . W_sc + b_sc.   Never materializes hidden.
// A panel async-copied to LDS once; all 8 waves read it via ds_load_b128.
// =====================================================================
__global__ void __launch_bounds__(256)
scores_kernel(const u16* __restrict__ gru_bf, const u16* __restrict__ Wg,
              const float* __restrict__ tvec, const float* __restrict__ Wsc,
              const float* __restrict__ b_sc, float* __restrict__ scores) {
  __shared__ __align__(16) u16 At[16 * D_];   // 16 KB gru panel
  __shared__ float ssum[16];
  const int m0  = blockIdx.x * 16;
  const int tid = threadIdx.x;
  {
    const u32 lds_base = (u32)(uintptr_t)At;
    const u16* gsrc = gru_bf + (size_t)m0 * D_;
#pragma unroll
    for (int i = 0; i < 4; ++i) {
      int j = tid + i * 256;
      async_copy_b128(lds_base + j * 16, gsrc + j * 8);
    }
    wait_async_all();
  }
  if (tid < 16) ssum[tid] = 0.f;
  __syncthreads();

  const int wave = tid >> 5, lane = tid & 31;
  const int idx = lane & 15, half = lane >> 4;
  v8f acc[2] = {};
  for (int k = 0; k < 16; ++k) {
    v16bf a = load_frag(At, D_, 0, k * 32);
#pragma unroll
    for (int q = 0; q < 2; ++q) {
      int f0 = (wave * 2 + q) * 16;
      acc[q] = wmma_bf16(a, load_frag(Wg, D_, f0, k * 32), acc[q]);
    }
  }
#pragma unroll
  for (int q = 0; q < 2; ++q) {
    int f = (wave * 2 + q) * 16 + idx;
    float w = Wsc[f];
#pragma unroll
    for (int r = 0; r < 8; ++r) {
      int mr = r + half * 8;
      int m = m0 + mr;
      int b = m / S_;                       // flat (b,s) row -> batch index
      float x = acc[q][r] + tvec[(size_t)b * F_ + f];
      float g = 0.5f * x * (1.f + erff(x * 0.70710678118654752f));
      atomicAdd(&ssum[mr], g * w);          // ds_add_f32
    }
  }
  __syncthreads();
  if (tid < 16) scores[m0 + tid] = ssum[tid] + b_sc[0];
}

// =====================================================================
// Kernel 5: per-batch softmax over S + weighted sum of gru_out + concat.
// =====================================================================
__global__ void __launch_bounds__(256)
softmax_interest_kernel(const float* __restrict__ scores, const u16* __restrict__ gru_bf,
                        const float* __restrict__ target, float* __restrict__ out) {
  __shared__ float red[256];
  __shared__ float wgt[S_];
  const int b = blockIdx.x;
  const int tid = threadIdx.x;
  float v = (tid < S_) ? scores[(size_t)b * S_ + tid] : -3.4e38f;
  red[tid] = v; __syncthreads();
  for (int o = 128; o > 0; o >>= 1) {
    if (tid < o) red[tid] = fmaxf(red[tid], red[tid + o]);
    __syncthreads();
  }
  float mx = red[0]; __syncthreads();
  float e = (tid < S_) ? __expf(v - mx) : 0.f;
  red[tid] = e; __syncthreads();
  for (int o = 128; o > 0; o >>= 1) {
    if (tid < o) red[tid] += red[tid + o];
    __syncthreads();
  }
  float inv = 1.f / red[0];
  if (tid < S_) wgt[tid] = e * inv;
  __syncthreads();
#pragma unroll
  for (int dd = 0; dd < 2; ++dd) {
    int d = tid + dd * 256;
    float accv = 0.f;
    for (int s = 0; s < S_; ++s)
      accv += wgt[s] * bf2f(gru_bf[((size_t)b * S_ + s) * D_ + d]);
    out[(size_t)b * (2 * D_) + d]      = accv;
    out[(size_t)b * (2 * D_) + D_ + d] = target[(size_t)b * D_ + d];
  }
}

// =====================================================================
extern "C" void kernel_launch(void* const* d_in, const int* in_sizes, int n_in,
                              void* d_out, int out_size, void* d_ws, size_t ws_size,
                              hipStream_t stream) {
  const float* target  = (const float*)d_in[0];
  const float* history = (const float*)d_in[1];
  const float* W_ih    = (const float*)d_in[2];
  const float* W_hh    = (const float*)d_in[3];
  const float* b_ih    = (const float*)d_in[4];
  const float* b_hh    = (const float*)d_in[5];
  const float* W_fc    = (const float*)d_in[6];
  const float* b_fc    = (const float*)d_in[7];
  const float* W_sc    = (const float*)d_in[8];
  const float* b_sc    = (const float*)d_in[9];
  float* out = (float*)d_out;

  // workspace carve-out (~846 MB)
  char* ws = (char*)d_ws;
  size_t off = 0;
  auto carve = [&](size_t bytes) -> void* {
    void* p = ws + off;
    off = (off + bytes + 255) & ~(size_t)255;
    return p;
  };
  u16*   gx     = (u16*)  carve((size_t)B_ * S_ * G_ * 2);
  u16*   gru_bf = (u16*)  carve((size_t)B_ * S_ * D_ * 2);
  u16*   Wih_bf = (u16*)  carve((size_t)G_ * D_ * 2);
  u16*   Whh_bf = (u16*)  carve((size_t)G_ * D_ * 2);
  u16*   Wg_bf  = (u16*)  carve((size_t)F_ * D_ * 2);
  u16*   Wt_bf  = (u16*)  carve((size_t)F_ * D_ * 2);
  u16*   tgt_bf = (u16*)  carve((size_t)B_ * D_ * 2);
  float* tvec   = (float*)carve((size_t)B_ * F_ * 4);
  float* h_f32  = (float*)carve((size_t)B_ * D_ * 4);
  u16*   h_bf16 = (u16*)  carve((size_t)B_ * D_ * 2);
  float* scores = (float*)carve((size_t)B_ * S_ * 4);

  dim3 blk(256);
  conv_bf16_kernel<<<dim3((G_ * D_ + 255) / 256), blk, 0, stream>>>(W_ih, Wih_bf, G_, D_, D_, 0);
  conv_bf16_kernel<<<dim3((G_ * D_ + 255) / 256), blk, 0, stream>>>(W_hh, Whh_bf, G_, D_, D_, 0);
  conv_bf16_kernel<<<dim3((F_ * D_ + 255) / 256), blk, 0, stream>>>(W_fc, Wg_bf, F_, D_, 2 * D_, 0);
  conv_bf16_kernel<<<dim3((F_ * D_ + 255) / 256), blk, 0, stream>>>(W_fc, Wt_bf, F_, D_, 2 * D_, D_);
  conv_bf16_kernel<<<dim3((B_ * D_ + 255) / 256), blk, 0, stream>>>(target, tgt_bf, B_, D_, D_, 0);
  zero_h_kernel<<<dim3((B_ * D_ + 255) / 256), blk, 0, stream>>>(h_f32, h_bf16, B_ * D_);

  gemm_gx_kernel<<<dim3(B_ * S_ / 16), blk, 0, stream>>>(history, Wih_bf, b_ih, gx);
  tvec_kernel<<<dim3(B_ / 16), blk, 0, stream>>>(tgt_bf, Wt_bf, b_fc, tvec);

  for (int t = 0; t < S_; ++t)
    gru_step_kernel<<<dim3(B_ / 16), blk, 0, stream>>>(gx, Whh_bf, b_hh, h_f32, h_bf16, gru_bf, t);

  scores_kernel<<<dim3(B_ * S_ / 16), blk, 0, stream>>>(gru_bf, Wg_bf, tvec, W_sc, b_sc, scores);
  softmax_interest_kernel<<<dim3(B_), blk, 0, stream>>>(scores, gru_bf, target, out);

  (void)in_sizes; (void)n_in; (void)out_size; (void)ws_size;
}
